// DeeperGCN_32796370272632
// MI455X (gfx1250) — compile-verified, matrix operand order checked by reading, not run on hardware
//
#include <hip/hip_runtime.h>
#include <hip/hip_bf16.h>

// ---------------------------------------------------------------------------
// DeeperGCN forward for MI455X (gfx1250, wave32, WMMA).
// Layers 4..6 of the reference are dead (only out[0..3] + h_init are returned)
// and are skipped. All dense linears go through v_wmma_f32_16x16x32_bf16.
// ---------------------------------------------------------------------------

typedef __attribute__((ext_vector_type(16))) __bf16 bf16x16;
typedef __attribute__((ext_vector_type(8)))  float  f32x8;

#define DD 256
#define GRAPHS 3000

__device__ __forceinline__ __bf16 f2bf(float f) { return (__bf16)f; }

// k-within-32-tile for fragment element e of a lane:
//   kwt = (e<8) ? (lane>>4)*8 + e : 16 + (lane>>4)*8 + (e-8)

// --- convert an f32 [256 x ncols] weight into WMMA-fragment-ordered bf16 ----
__global__ void wfrag_kernel(const float* __restrict__ w,
                             __bf16* __restrict__ dst, int ncols) {
  int idx = blockIdx.x * blockDim.x + threadIdx.x;
  int total = 256 * ncols;
  if (idx >= total) return;
  int k = idx / ncols, n = idx % ncols;
  int kt = k >> 5, r = k & 31;
  int half = r >> 4;
  int r2 = r & 15;
  int lane = (r2 >> 3) * 16 + (n & 15);
  int e = half * 8 + (r2 & 7);
  int ct = n >> 4;
  int nt = ncols >> 4;
  dst[(((kt * nt + ct) * 32) + lane) * 16 + e] = f2bf(w[idx]);
}

// --- WMMA GEMM: out[16 rows x NT*16 cols] = (A1(+A2)) @ Wfrag + bias (+res) -
template <int TPW>  // tiles per wave; block = 4 waves, NT = 4*TPW col tiles
__global__ __launch_bounds__(128) void gemm_wmma(
    const float* __restrict__ A1, const float* __restrict__ A2,
    const __bf16* __restrict__ Wf, const float* __restrict__ bias,
    const float* __restrict__ res, float* __restrict__ out,
    int nrows, int ldout) {
  constexpr int NT = 4 * TPW;
  __shared__ __align__(32) __bf16 ldsA[8 * 32 * 16];  // 16x256 strip, frag order

  const int t = threadIdx.x;
  const int rbase = blockIdx.x * 16;

  // Stage A strip (fused A1+A2, f32 -> bf16) into LDS in fragment order.
  for (int idx = t; idx < 16 * 256; idx += 128) {
    int m = idx >> 8, k = idx & 255;
    float v = 0.f;
    int row = rbase + m;
    if (row < nrows) {
      v = A1[(size_t)row * DD + k];
      if (A2) v += A2[(size_t)row * DD + k];
    }
    int kt = k >> 5, r = k & 31;
    int half = r >> 4, r2 = r & 15;
    int lane = (r2 >> 3) * 16 + m;
    int e = half * 8 + (r2 & 7);
    ldsA[((kt * 32 + lane) << 4) + e] = f2bf(v);
  }
  __syncthreads();

  const int wave = t >> 5, lane = t & 31;
  f32x8 zero = {};
  f32x8 acc[TPW];
#pragma unroll
  for (int j = 0; j < TPW; ++j) acc[j] = zero;

  const bf16x16* Wfv = reinterpret_cast<const bf16x16*>(Wf);
#pragma unroll
  for (int kt = 0; kt < 8; ++kt) {
    bf16x16 av = *reinterpret_cast<const bf16x16*>(&ldsA[(kt * 32 + lane) << 4]);
#pragma unroll
    for (int j = 0; j < TPW; ++j) {
      int ct = wave * TPW + j;
      bf16x16 bv = Wfv[(kt * NT + ct) * 32 + lane];
      acc[j] = __builtin_amdgcn_wmma_f32_16x16x32_bf16(
          false, av, false, bv, (short)0, acc[j], false, false);
    }
  }

  // Epilogue: C/D layout -> row = rbase + i + 8*(lane>=16), col = ct*16+(lane&15)
  const int rowoff = 8 * (lane >> 4);
  const int nlo = lane & 15;
#pragma unroll
  for (int j = 0; j < TPW; ++j) {
    int col = (wave * TPW + j) * 16 + nlo;
    float bcol = bias[col];
#pragma unroll
    for (int i = 0; i < 8; ++i) {
      int row = rbase + rowoff + i;
      if (row < nrows) {
        float v = acc[j][i] + bcol;
        if (res) v += res[(size_t)row * DD + col];
        out[(size_t)row * ldout + col] = v;
      }
    }
  }
}

// --- atom encoder: h = sum_f atom_emb[f, x[n,f], :] (+ vn_emb); h_init out ---
__global__ void encode_kernel(const int* __restrict__ x,
                              const float* __restrict__ atom,
                              const float* __restrict__ vn_emb,
                              float* __restrict__ h,
                              float* __restrict__ h_init_out) {
  int n = blockIdx.x, d = threadIdx.x;
  float s = 0.f;
#pragma unroll
  for (int f = 0; f < 9; ++f) {
    int v = x[n * 9 + f];
    s += atom[((size_t)(f * 64 + v)) * DD + d];
  }
  h_init_out[(size_t)n * DD + d] = s;
  h[(size_t)n * DD + d] = s + vn_emb[d];
}

__global__ void vninit_kernel(const float* __restrict__ vn_emb,
                              float* __restrict__ vn) {
  vn[(size_t)blockIdx.x * DD + threadIdx.x] = vn_emb[threadIdx.x];
}

__global__ void zero_kernel(float* __restrict__ p, int n) {
  int i = blockIdx.x * blockDim.x + threadIdx.x;
  if (i < n) p[i] = 0.f;
}

__global__ void copy_kernel(const float* __restrict__ s, float* __restrict__ d,
                            int n) {
  int i = blockIdx.x * blockDim.x + threadIdx.x;
  if (i < n) d[i] = s[i];
}

// --- GENConv message + scatter-add (edge_emb recomputed from bond table) ----
__global__ void edge_kernel(const float* __restrict__ h,
                            const int* __restrict__ eidx,
                            const int* __restrict__ eattr,
                            const float* __restrict__ bond,
                            float* __restrict__ agg, int E) {
  int e = blockIdx.x, d = threadIdx.x;
  int s = eidx[e], dst = eidx[E + e];
  int a0 = eattr[e * 3 + 0], a1 = eattr[e * 3 + 1], a2 = eattr[e * 3 + 2];
  float em = bond[(size_t)a0 * DD + d] + bond[(size_t)(8 + a1) * DD + d] +
             bond[(size_t)(16 + a2) * DD + d];
  float m = fmaxf(h[(size_t)s * DD + d] + em, 0.f) + 1e-7f;
  atomicAdd(&agg[(size_t)dst * DD + d], m);
}

// --- LayerNorm(256) + ReLU, one block (8 waves) per row --------------------
__global__ void ln_relu_kernel(const float* __restrict__ in,
                               float* __restrict__ out,
                               const float* __restrict__ g,
                               const float* __restrict__ b) {
  int row = blockIdx.x, d = threadIdx.x;
  float x = in[(size_t)row * DD + d];
  float s = x, q = x * x;
#pragma unroll
  for (int o = 16; o > 0; o >>= 1) {
    s += __shfl_xor(s, o, 32);
    q += __shfl_xor(q, o, 32);
  }
  __shared__ float ss[8], sq[8];
  int wid = d >> 5, lane = d & 31;
  if (lane == 0) { ss[wid] = s; sq[wid] = q; }
  __syncthreads();
  if (wid == 0) {
    float s2 = (lane < 8) ? ss[lane] : 0.f;
    float q2 = (lane < 8) ? sq[lane] : 0.f;
#pragma unroll
    for (int o = 4; o > 0; o >>= 1) {
      s2 += __shfl_xor(s2, o, 32);
      q2 += __shfl_xor(q2, o, 32);
    }
    if (lane == 0) { ss[0] = s2; sq[0] = q2; }
  }
  __syncthreads();
  float mu = ss[0] * (1.f / 256.f);
  float var = sq[0] * (1.f / 256.f) - mu * mu;
  float y = (x - mu) * rsqrtf(var + 1e-5f) * g[d] + b[d];
  out[(size_t)row * DD + d] = fmaxf(y, 0.f);
}

// --- segment-sum of nodes into graphs (batch sorted; atomics hit L2) --------
__global__ void segsum_kernel(const float* __restrict__ h2,
                              const int* __restrict__ batch,
                              float* __restrict__ vt) {
  int n = blockIdx.x, d = threadIdx.x;
  atomicAdd(&vt[(size_t)batch[n] * DD + d], h2[(size_t)n * DD + d]);
}

__global__ void addvn_kernel(float* __restrict__ h2,
                             const float* __restrict__ vn,
                             const int* __restrict__ batch) {
  int n = blockIdx.x, d = threadIdx.x;
  h2[(size_t)n * DD + d] += vn[(size_t)batch[n] * DD + d];
}

// ---------------------------------------------------------------------------
extern "C" void kernel_launch(void* const* d_in, const int* in_sizes, int n_in,
                              void* d_out, int out_size, void* d_ws,
                              size_t ws_size, hipStream_t stream) {
  (void)n_in; (void)out_size; (void)ws_size;
  const int* x       = (const int*)d_in[0];
  const int* eattr   = (const int*)d_in[1];
  const int* eidx    = (const int*)d_in[2];
  const int* batch   = (const int*)d_in[3];
  const float* atom  = (const float*)d_in[4];
  const float* bond  = (const float*)d_in[5];
  const float* vn_emb= (const float*)d_in[6];
  const float* gcn_w = (const float*)d_in[7];
  const float* gcn_b = (const float*)d_in[8];
  const float* norm_g= (const float*)d_in[9];
  const float* norm_b= (const float*)d_in[10];
  const float* ffn_w = (const float*)d_in[11];
  const float* ffn_b = (const float*)d_in[12];
  const float* vn_w1 = (const float*)d_in[13];
  const float* vn_b1 = (const float*)d_in[14];
  const float* vn_g1 = (const float*)d_in[15];
  const float* vn_be1= (const float*)d_in[16];
  const float* vn_w2 = (const float*)d_in[17];
  const float* vn_b2 = (const float*)d_in[18];
  const float* vn_g2 = (const float*)d_in[19];
  const float* vn_be2= (const float*)d_in[20];

  const int N = in_sizes[0] / 9;
  const int E = in_sizes[1] / 3;
  float* out = (float*)d_out;

  char* ws = (char*)d_ws;
  size_t off = 0;
  auto alloc = [&](size_t bytes) -> char* {
    char* p = ws + off;
    off += (bytes + 255) & ~(size_t)255;
    return p;
  };
  float* h    = (float*)alloc((size_t)N * DD * 4);
  float* h2   = (float*)alloc((size_t)N * DD * 4);
  float* agg  = (float*)alloc((size_t)N * DD * 4);
  float* vn   = (float*)alloc((size_t)GRAPHS * DD * 4);
  float* vt   = (float*)alloc((size_t)GRAPHS * DD * 4);
  float* va   = (float*)alloc((size_t)GRAPHS * DD * 4);
  __bf16* gcnf = (__bf16*)alloc((size_t)4 * DD * DD * 2);
  __bf16* ffnf = (__bf16*)alloc((size_t)4 * DD * 64 * 2);
  __bf16* w1f  = (__bf16*)alloc((size_t)3 * DD * DD * 2);
  __bf16* w2f  = (__bf16*)alloc((size_t)3 * DD * DD * 2);

  // One-time weight conversion to WMMA-fragment bf16 (dead layers 4..6 skipped)
  for (int l = 0; l < 4; ++l) {
    wfrag_kernel<<<(DD * DD + 255) / 256, 256, 0, stream>>>(
        gcn_w + (size_t)l * DD * DD, gcnf + (size_t)l * DD * DD, DD);
    wfrag_kernel<<<(DD * 64 + 255) / 256, 256, 0, stream>>>(
        ffn_w + (size_t)l * DD * 64, ffnf + (size_t)l * DD * 64, 64);
  }
  for (int l = 0; l < 3; ++l) {
    wfrag_kernel<<<(DD * DD + 255) / 256, 256, 0, stream>>>(
        vn_w1 + (size_t)l * DD * DD, w1f + (size_t)l * DD * DD, DD);
    wfrag_kernel<<<(DD * DD + 255) / 256, 256, 0, stream>>>(
        vn_w2 + (size_t)l * DD * DD, w2f + (size_t)l * DD * DD, DD);
  }

  encode_kernel<<<N, 256, 0, stream>>>(x, atom, vn_emb, h, out + (size_t)N * DD);
  vninit_kernel<<<GRAPHS, 256, 0, stream>>>(vn_emb, vn);

  const int rtN = (N + 15) / 16;
  const int rtG = (GRAPHS + 15) / 16;
  const int ND = N * DD;

  // ---- layer 0: h = (h + agg(msg)) @ W0 + b0 ; out[:, 0:64] = h @ F0 + f0
  zero_kernel<<<(ND + 255) / 256, 256, 0, stream>>>(agg, ND);
  edge_kernel<<<E, 256, 0, stream>>>(h, eidx, eattr, bond, agg, E);
  gemm_wmma<4><<<rtN, 128, 0, stream>>>(h, agg, gcnf, gcn_b, nullptr, h, N, DD);
  gemm_wmma<1><<<rtN, 128, 0, stream>>>(h, nullptr, ffnf, ffn_b, nullptr, out,
                                        N, DD);

  // ---- layers 1..3 (4..6 are dead)
  for (int l = 1; l < 4; ++l) {
    const int j = l - 1;
    ln_relu_kernel<<<N, 256, 0, stream>>>(h, h2, norm_g + (size_t)j * DD,
                                          norm_b + (size_t)j * DD);
    copy_kernel<<<(GRAPHS * DD + 255) / 256, 256, 0, stream>>>(vn, vt,
                                                               GRAPHS * DD);
    segsum_kernel<<<N, 256, 0, stream>>>(h2, batch, vt);
    gemm_wmma<4><<<rtG, 128, 0, stream>>>(vt, nullptr, w1f + (size_t)j * DD * DD,
                                          vn_b1 + (size_t)j * DD, nullptr, va,
                                          GRAPHS, DD);
    ln_relu_kernel<<<GRAPHS, 256, 0, stream>>>(va, va, vn_g1 + (size_t)j * DD,
                                               vn_be1 + (size_t)j * DD);
    gemm_wmma<4><<<rtG, 128, 0, stream>>>(va, nullptr, w2f + (size_t)j * DD * DD,
                                          vn_b2 + (size_t)j * DD, nullptr, vn,
                                          GRAPHS, DD);
    ln_relu_kernel<<<GRAPHS, 256, 0, stream>>>(vn, vn, vn_g2 + (size_t)j * DD,
                                               vn_be2 + (size_t)j * DD);
    addvn_kernel<<<N, 256, 0, stream>>>(h2, vn, batch);

    zero_kernel<<<(ND + 255) / 256, 256, 0, stream>>>(agg, ND);
    edge_kernel<<<E, 256, 0, stream>>>(h2, eidx, eattr, bond, agg, E);
    gemm_wmma<4><<<rtN, 128, 0, stream>>>(h2, agg, gcnf + (size_t)l * DD * DD,
                                          gcn_b + (size_t)l * DD, h, h, N, DD);
    gemm_wmma<1><<<rtN, 128, 0, stream>>>(h, nullptr,
                                          ffnf + (size_t)l * DD * 64,
                                          ffn_b + (size_t)l * 64, nullptr,
                                          out + (size_t)l * 64, N, DD);
  }
}
